// Attention_8211977470138
// MI455X (gfx1250) — matrix operand order repaired and checked by reading, compiled, not run on hardware
//
#include <hip/hip_runtime.h>
#include <hip/hip_bf16.h>

// ---------------------------------------------------------------------------
// CDNA5 (gfx1250) fused PVT-style cross attention — all six GEMMs
// (im2col-conv, KV proj, Q proj, Q*K^T, P*V, out proj) on
// v_wmma_f32_16x16x32_f16; weight/K/V staging via global_load_async_to_lds.
//   kernel 0: one-time f32->f16 conversion of weights into ws
//   kernel 1: conv(stride4)-as-GEMM + LayerNorm + KV GEMM -> K / V^T in ws
//             (cross-batch swap applied at write time)
//   kernel 2: per 64-query-row tile: Q=X*Wq^T, S=Q*K^T, softmax, O=P*V,
//             OUT=O*Wp^T+bp
// ---------------------------------------------------------------------------

typedef __attribute__((ext_vector_type(16))) _Float16 v16h;
typedef __attribute__((ext_vector_type(8)))  _Float16 v8h;
typedef __attribute__((ext_vector_type(8)))  float    v8f;

#define DIM   128
#define NH    2
#define HD    64
#define NTOK  4096
#define NKV   256
#define BATCH 8

// ---- WMMA fragment loaders (wave32 layouts per CDNA5 ISA 7.12.2) ----------
static __device__ __forceinline__ v16h
load_a_frag(const _Float16* base, int lda, int row0, int k0, int lane) {
    const int sub = lane & 15;
    const int hi  = lane >> 4;
    const _Float16* p = base + (size_t)(row0 + sub) * lda + k0 + hi * 8;
    v8h lo = *(const v8h*)(p);
    v8h hh = *(const v8h*)(p + 16);
    v16h r;
#pragma unroll
    for (int i = 0; i < 8; ++i) { r[i] = lo[i]; r[8 + i] = hh[i]; }
    return r;
}

static __device__ __forceinline__ v16h
load_b_frag(const _Float16* bt, int ldb, int col0, int k0, int lane) {
    const int sub = lane & 15;
    const int hi  = lane >> 4;
    const _Float16* p = bt + (size_t)(col0 + sub) * ldb + k0 + hi * 8;
    v8h lo = *(const v8h*)(p);
    v8h hh = *(const v8h*)(p + 16);
    v16h r;
#pragma unroll
    for (int i = 0; i < 8; ++i) { r[i] = lo[i]; r[8 + i] = hh[i]; }
    return r;
}

static __device__ __forceinline__ v8f
wmma_f16(v16h a, v16h b, v8f c) {
    return __builtin_amdgcn_wmma_f32_16x16x32_f16(false, a, false, b,
                                                  (short)0, c, false, false);
}

// Async memory->LDS DMA (CDNA5, ASYNCcnt-tracked; bypasses VGPRs).
static __device__ __forceinline__ void
async_copy_b128(void* lds_dst, const void* gbase, unsigned lds_off, unsigned g_off) {
    unsigned lds_addr = (unsigned)(uintptr_t)lds_dst + lds_off;
    asm volatile("global_load_async_to_lds_b128 %0, %1, %2"
                 :: "v"(lds_addr), "v"(g_off), "s"(gbase)
                 : "memory");
}

static __device__ __forceinline__ void async_wait0() {
    asm volatile("s_wait_asynccnt 0x0" ::: "memory");
}

// ---------------------------------------------------------------------------
// Kernel 0: elementwise f32 -> f16
// ---------------------------------------------------------------------------
__global__ __launch_bounds__(256, 1)
void cvt_f16(const float* __restrict__ src, _Float16* __restrict__ dst, int n) {
    int i = blockIdx.x * 256 + threadIdx.x;
    if (i < n) dst[i] = (_Float16)src[i];
}

// ---------------------------------------------------------------------------
// Kernel 1: im2col conv GEMM + LN + KV GEMM.  One block per (b, 32 pixels).
// 256 threads = 8 waves.  A = patches [32 x 2048], B = sr_w16 rows (already
// column-contiguous for load_b_frag).  K tiled in 256-element chunks.
// ---------------------------------------------------------------------------
__global__ __launch_bounds__(256, 1)
void pvt_kv_prep(const float* __restrict__ x,
                 const _Float16* __restrict__ srw16,  // [128][2048] f16
                 const float* __restrict__ sr_b,
                 const float* __restrict__ ln_w,
                 const float* __restrict__ ln_b,
                 const _Float16* __restrict__ wkv16,  // [256][128] f16
                 _Float16* __restrict__ Kws,          // [B][NH][256][64]
                 _Float16* __restrict__ Vtws) {       // [B][NH][64][256]
    __shared__ __align__(16) _Float16 Abuf[32 * 256];    // 16 KB  im2col chunk
    __shared__ __align__(16) _Float16 Bbuf[128 * 256];   // 64 KB  weights (conv / Wkv)
    __shared__ __align__(16) float    convout[32 * 128]; // 16 KB
    __shared__ __align__(16) _Float16 ybuf[32 * 128];    //  8 KB  LN output
    __shared__ float srb_l[128];

    const int tid  = threadIdx.x;
    const int lane = tid & 31;
    const int wv   = tid >> 5;
    const int b    = blockIdx.x >> 3;
    const int p0   = (blockIdx.x & 7) * 32;

    if (tid < 128) srb_l[tid] = sr_b[tid];

    v8f acc[2] = {};

    for (int c0 = 0; c0 < 2048; c0 += 256) {
        // async-DMA the 128x256 f16 weight chunk (row o: 256 contiguous halves)
        const void* Bg = (const char*)srw16 + 2u * (unsigned)c0;
        for (int e = tid; e < 4096; e += 256)
            async_copy_b128(Bbuf, Bg, (unsigned)e * 16u,
                            ((unsigned)(e >> 5) * 4096u) + ((unsigned)(e & 31) * 16u));

        // im2col gather of A chunk (overlaps the DMA): kidx = rc*128 + i
        for (int e = tid; e < 32 * 256; e += 256) {
            int pl = e >> 8, kk = e & 255;
            int kidx = c0 + kk;
            int rc = kidx >> 7, i = kidx & 127;
            int r = rc >> 2, cc = rc & 3;
            int pix = p0 + pl;
            int ph = pix >> 4, pw = pix & 15;
            int n = (4 * ph + r) * 64 + 4 * pw + cc;
            Abuf[pl * 256 + kk] = (_Float16)x[((size_t)b * NTOK + n) * DIM + i];
        }
        async_wait0();
        __syncthreads();

        // accumulate: 16 tiles (2 mt x 8 nt), 2 per wave
#pragma unroll
        for (int i = 0; i < 2; ++i) {
            const int t = wv * 2 + i, mt = t >> 3, nt = t & 7;
#pragma unroll
            for (int k0 = 0; k0 < 256; k0 += 32) {
                v16h a  = load_a_frag(Abuf, 256, mt * 16, k0, lane);
                v16h bb = load_b_frag(Bbuf, 256, nt * 16, k0, lane);
                acc[i] = wmma_f16(a, bb, acc[i]);
            }
        }
        __syncthreads();
    }

    // prefetch Wkv (64 KB contiguous) into Bbuf while LN runs
    for (int e = tid; e < 4096; e += 256)
        async_copy_b128(Bbuf, wkv16, (unsigned)e * 16u, (unsigned)e * 16u);

    // conv result + bias -> LDS f32
#pragma unroll
    for (int i = 0; i < 2; ++i) {
        const int t = wv * 2 + i, mt = t >> 3, nt = t & 7;
        const int col = nt * 16 + (lane & 15), rbase = mt * 16 + ((lane >> 4) << 3);
#pragma unroll
        for (int r = 0; r < 8; ++r)
            convout[(rbase + r) * 128 + col] = acc[i][r] + srb_l[col];
    }
    __syncthreads();

    // LayerNorm: wave w owns rows [4w, 4w+4); lane covers 4 channels
    for (int rr = wv * 4; rr < wv * 4 + 4; ++rr) {
        const float* row = &convout[rr * 128];
        float v4[4];
        float ssum = 0.f;
#pragma unroll
        for (int j = 0; j < 4; ++j) { v4[j] = row[lane * 4 + j]; ssum += v4[j]; }
        for (int off = 16; off > 0; off >>= 1) ssum += __shfl_xor(ssum, off);
        const float mu = ssum * (1.0f / 128.0f);
        float dsum = 0.f;
#pragma unroll
        for (int j = 0; j < 4; ++j) { v4[j] -= mu; dsum += v4[j] * v4[j]; }
        for (int off = 16; off > 0; off >>= 1) dsum += __shfl_xor(dsum, off);
        const float rs = rsqrtf(dsum * (1.0f / 128.0f) + 1e-5f);
#pragma unroll
        for (int j = 0; j < 4; ++j) {
            const int col = lane * 4 + j;
            ybuf[rr * 128 + col] = (_Float16)(v4[j] * rs * ln_w[col] + ln_b[col]);
        }
    }
    async_wait0();
    __syncthreads();

    // KV GEMM: [32 x 128] * [128 x 256] -> 32 tiles (2 mt x 16 nt), 4 per wave
    const int bd = (b + 4) & 7;   // cross-batch swap applied on store
#pragma unroll
    for (int i = 0; i < 4; ++i) {
        const int t = wv * 4 + i, mt = t >> 4, nt = t & 15;
        v8f a2 = {};
#pragma unroll
        for (int k0 = 0; k0 < 128; k0 += 32) {
            v16h a  = load_a_frag(ybuf, 128, mt * 16, k0, lane);
            v16h bb = load_b_frag(Bbuf, 128, nt * 16, k0, lane);
            a2 = wmma_f16(a, bb, a2);
        }
        const int col   = nt * 16 + (lane & 15);
        const int rbase = mt * 16 + ((lane >> 4) << 3);
        if (col < 128) {            // K output channel (uniform per tile)
            const int head = col >> 6, dd = col & 63;
#pragma unroll
            for (int r = 0; r < 8; ++r) {
                const int pix = p0 + rbase + r;
                Kws[((size_t)(bd * NH + head) * NKV + pix) * HD + dd] = (_Float16)a2[r];
            }
        } else {                    // V output channel -> transposed store
            const int c2 = col - 128, head = c2 >> 6, dd = c2 & 63;
#pragma unroll
            for (int r = 0; r < 8; ++r) {
                const int pix = p0 + rbase + r;
                Vtws[((size_t)(bd * NH + head) * HD + dd) * NKV + pix] = (_Float16)a2[r];
            }
        }
    }
}

// ---------------------------------------------------------------------------
// Kernel 2: fused Q / attention / projection.  One block per (b, 64 rows).
// ---------------------------------------------------------------------------
__global__ __launch_bounds__(256, 1)
void pvt_attn(const float* __restrict__ x,
              const _Float16* __restrict__ wq16,   // [128][128] f16
              const _Float16* __restrict__ wp16,   // [128][128] f16
              const float* __restrict__ bp,
              const _Float16* __restrict__ Kws,
              const _Float16* __restrict__ Vtws,
              float* __restrict__ out) {
    __shared__ __align__(16) _Float16 bufA [64 * 128];   // x tile -> O tile
    __shared__ __align__(16) _Float16 bufB [128 * 128];  // Wq -> S/P -> Wp
    __shared__ __align__(16) _Float16 qbuf [64 * 128];   // Q (scaled, f16)
    __shared__ __align__(16) _Float16 kbuf [NKV * HD];   // K, current head
    __shared__ __align__(16) _Float16 vtbuf[HD * NKV];   // V^T, current head
    __shared__ float rinv[64];
    __shared__ float bp_l[128];

    const int tid  = threadIdx.x;
    const int lane = tid & 31;
    const int wv   = tid >> 5;
    const int b    = blockIdx.x >> 6;
    const int n0   = (blockIdx.x & 63) * 64;

    // ---- kick off head-0 K/V + Wq DMA before doing anything else ----------
    {
        const void* Kg = (const void*)(Kws  + (size_t)(b * NH + 0) * NKV * HD);
        const void* Vg = (const void*)(Vtws + (size_t)(b * NH + 0) * HD * NKV);
        for (int e = tid; e < 2048; e += 256) {        // 2048 x 16B = 32 KB each
            async_copy_b128(kbuf,  Kg, (unsigned)e * 16u, (unsigned)e * 16u);
            async_copy_b128(vtbuf, Vg, (unsigned)e * 16u, (unsigned)e * 16u);
            async_copy_b128(bufB, wq16, (unsigned)e * 16u, (unsigned)e * 16u);
        }
    }

    // ---- phase 1: x tile f32->f16 (overlaps the async DMA) ----------------
    const float* xg = x + ((size_t)b * NTOK + n0) * DIM;
    for (int e = tid; e < 64 * 128; e += 256) bufA[e] = (_Float16)xg[e];
    if (tid < 128) bp_l[tid] = bp[tid];
    async_wait0();
    __syncthreads();

    // ---- phase 2: Q = X * Wq^T  (scale folded into f16 convert) -----------
    const float scale = 0.125f;  // hd^-0.5
    for (int t = wv; t < 32; t += 8) {
        const int mt = t >> 3, nt = t & 7;
        v8f acc = {};
#pragma unroll
        for (int k0 = 0; k0 < 128; k0 += 32) {
            v16h a  = load_a_frag(bufA, 128, mt * 16, k0, lane);
            v16h bb = load_b_frag(bufB, 128, nt * 16, k0, lane);
            acc = wmma_f16(a, bb, acc);
        }
        const int col   = nt * 16 + (lane & 15);
        const int rbase = mt * 16 + ((lane >> 4) << 3);
#pragma unroll
        for (int r = 0; r < 8; ++r)
            qbuf[(rbase + r) * 128 + col] = (_Float16)(acc[r] * scale);
    }
    __syncthreads();

    // ---- per-head flash attention (K/V fully LDS-resident) ----------------
    for (int h = 0; h < NH; ++h) {
        // S = Q_h * K^T : M=64, N=256, K=64.  K rows are B-columns directly.
        for (int i = 0; i < 8; ++i) {
            const int t  = wv * 8 + i;
            const int mt = t >> 4, nt = t & 15;
            v8f acc = {};
#pragma unroll
            for (int k0 = 0; k0 < 64; k0 += 32) {
                v16h a  = load_a_frag(qbuf + h * HD, 128, mt * 16, k0, lane);
                v16h bb = load_b_frag(kbuf, HD, nt * 16, k0, lane);
                acc = wmma_f16(a, bb, acc);
            }
            const int col   = nt * 16 + (lane & 15);
            const int rbase = mt * 16 + ((lane >> 4) << 3);
#pragma unroll
            for (int r = 0; r < 8; ++r)
                bufB[(rbase + r) * 256 + col] = (_Float16)acc[r];
        }
        __syncthreads();

        // softmax: wave w owns rows [8w, 8w+8); unnormalized exp kept in P,
        // 1/rowsum folded into O accumulators later.
        for (int r = wv * 8; r < wv * 8 + 8; ++r) {
            v8h* pr = (v8h*)&bufB[r * 256 + lane * 8];
            v8h pv = *pr;
            float s[8];
            float mx = -1e30f;
#pragma unroll
            for (int j = 0; j < 8; ++j) { s[j] = (float)pv[j]; mx = fmaxf(mx, s[j]); }
            for (int off = 16; off > 0; off >>= 1) mx = fmaxf(mx, __shfl_xor(mx, off));
            float sum = 0.f;
#pragma unroll
            for (int j = 0; j < 8; ++j) { s[j] = __expf(s[j] - mx); sum += s[j]; }
            for (int off = 16; off > 0; off >>= 1) sum += __shfl_xor(sum, off);
            v8h ov;
#pragma unroll
            for (int j = 0; j < 8; ++j) ov[j] = (_Float16)s[j];
            *pr = ov;
            if (lane == 0) rinv[r] = 1.0f / sum;
        }
        __syncthreads();

        // O_h = P * V : M=64, N=64, K=256.  V^T rows are B-columns.
        for (int i = 0; i < 2; ++i) {
            const int t  = wv * 2 + i;
            const int mt = t >> 2, nt = t & 3;
            v8f acc = {};
#pragma unroll
            for (int k0 = 0; k0 < 256; k0 += 32) {
                v16h a  = load_a_frag(bufB, 256, mt * 16, k0, lane);
                v16h bb = load_b_frag(vtbuf, 256, nt * 16, k0, lane);
                acc = wmma_f16(a, bb, acc);
            }
            const int col   = h * HD + nt * 16 + (lane & 15);
            const int rbase = mt * 16 + ((lane >> 4) << 3);
#pragma unroll
            for (int r = 0; r < 8; ++r) {
                const int row = rbase + r;
                bufA[row * 128 + col] = (_Float16)(acc[r] * rinv[row]);
            }
        }
        __syncthreads();   // P (bufB) fully consumed; kbuf/vtbuf reusable

        // prefetch next head's K/V
        if (h + 1 < NH) {
            const void* Kg = (const void*)(Kws  + (size_t)(b * NH + h + 1) * NKV * HD);
            const void* Vg = (const void*)(Vtws + (size_t)(b * NH + h + 1) * HD * NKV);
            for (int e = tid; e < 2048; e += 256) {
                async_copy_b128(kbuf,  Kg, (unsigned)e * 16u, (unsigned)e * 16u);
                async_copy_b128(vtbuf, Vg, (unsigned)e * 16u, (unsigned)e * 16u);
            }
            async_wait0();
            __syncthreads();
        }
    }

    // ---- phase 4: OUT = O * Wp^T + bp -------------------------------------
    for (int e = tid; e < 2048; e += 256)
        async_copy_b128(bufB, wp16, (unsigned)e * 16u, (unsigned)e * 16u);
    async_wait0();
    __syncthreads();

    float* og = out + ((size_t)b * NTOK + n0) * DIM;
    for (int t = wv; t < 32; t += 8) {
        const int mt = t >> 3, nt = t & 7;
        v8f acc = {};
#pragma unroll
        for (int k0 = 0; k0 < 128; k0 += 32) {
            v16h a  = load_a_frag(bufA, 128, mt * 16, k0, lane);
            v16h bb = load_b_frag(bufB, 128, nt * 16, k0, lane);
            acc = wmma_f16(a, bb, acc);
        }
        const int col   = nt * 16 + (lane & 15);
        const int rbase = mt * 16 + ((lane >> 4) << 3);
#pragma unroll
        for (int r = 0; r < 8; ++r)
            og[(size_t)(rbase + r) * DIM + col] = acc[r] + bp_l[col];
    }
}

// ---------------------------------------------------------------------------
extern "C" void kernel_launch(void* const* d_in, const int* in_sizes, int n_in,
                              void* d_out, int out_size, void* d_ws, size_t ws_size,
                              hipStream_t stream) {
    const float* x    = (const float*)d_in[0];
    const float* Wq   = (const float*)d_in[1];
    const float* Wkv  = (const float*)d_in[2];
    const float* sr_w = (const float*)d_in[3];
    const float* sr_b = (const float*)d_in[4];
    const float* ln_w = (const float*)d_in[5];
    const float* ln_b = (const float*)d_in[6];
    const float* Wp   = (const float*)d_in[7];
    const float* bp   = (const float*)d_in[8];
    float* out        = (float*)d_out;

    _Float16* ws    = (_Float16*)d_ws;
    _Float16* Kws   = ws;              // 262144 halves (512 KB)
    _Float16* Vtws  = ws + 262144;     // 262144 halves
    _Float16* srw16 = ws + 524288;     // 262144 halves
    _Float16* wkv16 = ws + 786432;     //  32768 halves
    _Float16* wq16  = ws + 819200;     //  16384 halves
    _Float16* wp16  = ws + 835584;     //  16384 halves  (total ~1.7 MB)

    cvt_f16<<<1024, 256, 0, stream>>>(sr_w, srw16, 262144);
    cvt_f16<<<128,  256, 0, stream>>>(Wkv,  wkv16, 32768);
    cvt_f16<<<64,   256, 0, stream>>>(Wq,   wq16,  16384);
    cvt_f16<<<64,   256, 0, stream>>>(Wp,   wp16,  16384);

    pvt_kv_prep<<<BATCH * 8, 256, 0, stream>>>(x, srw16, sr_b, ln_w, ln_b,
                                               wkv16, Kws, Vtws);
    pvt_attn<<<BATCH * (NTOK / 64), 256, 0, stream>>>(x, wq16, wp16, bp,
                                                      Kws, Vtws, out);
}
